// NaiveSparseAttention_37142877176014
// MI455X (gfx1250) — matrix operand order repaired and checked by reading, not compiled, not run on hardware
//
#include <hip/hip_runtime.h>

typedef __attribute__((ext_vector_type(16))) _Float16 v16h;
typedef __attribute__((ext_vector_type(8)))  _Float16 v8h;
typedef __attribute__((ext_vector_type(8)))  float    v8f;

#define SCALE 0.08838834764831845f  // 128^-0.5

// ---------------------------------------------------------------------------
// Elementwise f32 -> f16
// ---------------------------------------------------------------------------
__global__ __launch_bounds__(256) void cvt_f32_to_f16(const float* __restrict__ in,
                                                      _Float16* __restrict__ out, int n) {
  int t = blockIdx.x * 256 + threadIdx.x;
  if (t < n) out[t] = (_Float16)in[t];
}

// W[K][N] f32 -> Wt[N][K] f16 (transposed so WMMA B-operand loads are contiguous)
__global__ __launch_bounds__(256) void cvt_wt_kernel(const float* __restrict__ W,
                                                     _Float16* __restrict__ Wt, int K, int N) {
  int t = blockIdx.x * 256 + threadIdx.x;
  int kw = K >> 3;
  if (t >= N * kw) return;
  int n = t / kw;
  int k0 = (t % kw) << 3;
#pragma unroll
  for (int i = 0; i < 8; ++i)
    Wt[(size_t)n * K + k0 + i] = (_Float16)W[(size_t)(k0 + i) * N + n];
}

// ---------------------------------------------------------------------------
// Generic WMMA GEMM: C[M][N] = A[M][K](f16,row) @ Bt[N][K](f16, B transposed)
// One wave computes a 16x64 tile of C. 256 threads = 8 waves per block.
// ---------------------------------------------------------------------------
__global__ __launch_bounds__(256) void gemm_f16_16x64(const _Float16* __restrict__ A,
                                                      const _Float16* __restrict__ Bt,
                                                      float* __restrict__ C,
                                                      int M, int N, int K) {
  int wid  = blockIdx.x * 8 + (threadIdx.x >> 5);
  int lane = threadIdx.x & 31;
  int r = lane & 15, hi = lane >> 4;
  int tilesN = N >> 6;
  int tm = (wid / tilesN) << 4;
  int tn = (wid % tilesN) << 6;
  if (tm >= M) return;

  v8f a0 = {0.f,0.f,0.f,0.f,0.f,0.f,0.f,0.f};
  v8f a1 = a0, a2 = a0, a3 = a0;

  const _Float16* arow  = A  + (size_t)(tm + r) * K;
  const _Float16* brow0 = Bt + (size_t)(tn + r) * K + hi * 16;
  const _Float16* brow1 = brow0 + (size_t)16 * K;
  const _Float16* brow2 = brow1 + (size_t)16 * K;
  const _Float16* brow3 = brow2 + (size_t)16 * K;

  for (int kc = 0; kc < K; kc += 32) {
    v16h Ao;
    ((v8h*)&Ao)[0] = *(const v8h*)(arow + kc + hi * 16);       // e0..7  : K = hi*16+0..7
    ((v8h*)&Ao)[1] = *(const v8h*)(arow + kc + hi * 16 + 8);   // e8..15 : K = hi*16+8..15
    v16h B0 = *(const v16h*)(brow0 + kc);
    v16h B1 = *(const v16h*)(brow1 + kc);
    v16h B2 = *(const v16h*)(brow2 + kc);
    v16h B3 = *(const v16h*)(brow3 + kc);
    a0 = __builtin_amdgcn_wmma_f32_16x16x32_f16(false, Ao, false, B0, (short)0, a0, false, false);
    a1 = __builtin_amdgcn_wmma_f32_16x16x32_f16(false, Ao, false, B1, (short)0, a1, false, false);
    a2 = __builtin_amdgcn_wmma_f32_16x16x32_f16(false, Ao, false, B2, (short)0, a2, false, false);
    a3 = __builtin_amdgcn_wmma_f32_16x16x32_f16(false, Ao, false, B3, (short)0, a3, false, false);
  }
#pragma unroll
  for (int j = 0; j < 8; ++j) {
    int m = tm + j + 8 * hi;                     // C/D layout: VGPR j -> M=j (+8 for hi lanes)
    float* cp = C + (size_t)m * N + tn + r;
    cp[0] = a0[j]; cp[16] = a1[j]; cp[32] = a2[j]; cp[48] = a3[j];
  }
}

// ---------------------------------------------------------------------------
// Gates: g = sigmoid(x @ Wg), x f32 [M][K], Wg f32 [K][48]
// ---------------------------------------------------------------------------
__global__ __launch_bounds__(256) void gate_kernel(const float* __restrict__ x,
                                                   const float* __restrict__ Wg,
                                                   float* __restrict__ g) {
  int t = blockIdx.x * 256 + threadIdx.x;       // 2048*48
  if (t >= 2048 * 48) return;
  int m = t / 48, j = t % 48;
  float s = 0.f;
  for (int k = 0; k < 2048; ++k)
    s += x[(size_t)m * 2048 + k] * Wg[(size_t)k * 48 + j];
  g[t] = 1.0f / (1.0f + __expf(-s));
}

// ---------------------------------------------------------------------------
// RoPE in-place on q,k (f32 [b][n][h][128]) + f16 copies
// ---------------------------------------------------------------------------
__global__ __launch_bounds__(256) void rope_kernel(float* __restrict__ q, float* __restrict__ k,
                                                   _Float16* __restrict__ qh,
                                                   _Float16* __restrict__ kh) {
  int t = blockIdx.x * 256 + threadIdx.x;       // 2*1024*16*64
  if (t >= 2 * 1024 * 16 * 64) return;
  int i = t & 63, h = (t >> 6) & 15, n = (t >> 10) & 1023, b = (t >> 20) & 1;
  float inv = __powf(10000.0f, -(float)i * (1.0f / 64.0f));
  float th = (float)n * inv;
  float sn, cs;
  __sincosf(th, &sn, &cs);
  size_t base = ((size_t)(b * 1024 + n) * 16 + h) << 7;
  float q1 = q[base + i], q2 = q[base + 64 + i];
  float k1 = k[base + i], k2 = k[base + 64 + i];
  float nq1 = q1 * cs - q2 * sn, nq2 = q1 * sn + q2 * cs;
  float nk1 = k1 * cs - k2 * sn, nk2 = k1 * sn + k2 * cs;
  q[base + i] = nq1; q[base + 64 + i] = nq2;
  k[base + i] = nk1; k[base + 64 + i] = nk2;
  qh[base + i] = (_Float16)nq1; qh[base + 64 + i] = (_Float16)nq2;
  kh[base + i] = (_Float16)nk1; kh[base + 64 + i] = (_Float16)nk2;
}

// v f32 [b][n][h][d] -> vt f16 [b][h][d][n]  (key-contiguous for PV B-operand)
__global__ __launch_bounds__(256) void repack_v_kernel(const float* __restrict__ v,
                                                       _Float16* __restrict__ vt) {
  int t = blockIdx.x * 256 + threadIdx.x;       // 2*16*128*1024
  if (t >= 2 * 16 * 128 * 1024) return;
  int n = t & 1023, d = (t >> 10) & 127, h = (t >> 17) & 15, b = (t >> 21) & 1;
  vt[t] = (_Float16)v[(((size_t)(b * 1024 + n) * 16 + h) << 7) + d];
}

// Block means of rope'd k and v -> k_cmp, v_cmp f32 [b][blk][h][d]
__global__ __launch_bounds__(256) void kv_cmp_kernel(const float* __restrict__ k,
                                                     const float* __restrict__ v,
                                                     float* __restrict__ kc,
                                                     float* __restrict__ vc) {
  int t = blockIdx.x * 256 + threadIdx.x;       // 2*16*16*128
  if (t >= 2 * 16 * 16 * 128) return;
  int d = t & 127, h = (t >> 7) & 15, blk = (t >> 11) & 15, b = (t >> 15) & 1;
  float sk = 0.f, sv = 0.f;
  for (int i = 0; i < 64; ++i) {
    size_t idx = (((size_t)(b * 1024 + blk * 64 + i) * 16 + h) << 7) + d;
    sk += k[idx]; sv += v[idx];
  }
  kc[t] = sk * (1.0f / 64.0f);
  vc[t] = sv * (1.0f / 64.0f);
}

// ---------------------------------------------------------------------------
// Compressed attention + exact top-8 block selection (nb=16, tiny -> VALU)
// ---------------------------------------------------------------------------
__global__ __launch_bounds__(256) void cmp_attn_kernel(const float* __restrict__ q,
                                                       const float* __restrict__ kc,
                                                       const float* __restrict__ vc,
                                                       float* __restrict__ ocmp,
                                                       unsigned* __restrict__ sel) {
  int t = blockIdx.x * 256 + threadIdx.x;       // 2*16*1024
  if (t >= 2 * 16 * 1024) return;
  int n = t & 1023, h = (t >> 10) & 15, b = (t >> 14) & 1;
  const float* qp = q + (((size_t)(b * 1024 + n) * 16 + h) << 7);

  float s[16];
  float mx = -1e30f;
  bool any = false;
#pragma unroll
  for (int j = 0; j < 16; ++j) {
    const float* kp = kc + (((size_t)(b * 16 + j) * 16 + h) << 7);
    float dot = 0.f;
    for (int d = 0; d < 128; ++d) dot += qp[d] * kp[d];
    s[j] = dot * SCALE;
    bool valid = n >= (j + 1) * 64 - 1;
    if (valid) { mx = fmaxf(mx, s[j]); any = true; }
  }
  float p[16]; float sum = 0.f;
#pragma unroll
  for (int j = 0; j < 16; ++j) {
    bool valid = n >= (j + 1) * 64 - 1;
    p[j] = valid ? __expf(s[j] - mx) : 0.f;
    sum += p[j];
  }
  float inv = any ? 1.0f / sum : 0.f;
#pragma unroll
  for (int j = 0; j < 16; ++j) p[j] *= inv;     // !any -> all zero (matches ref)

  float* op = ocmp + (((size_t)(b * 1024 + n) * 16 + h) << 7);
  for (int d = 0; d < 128; ++d) {
    float a = 0.f;
#pragma unroll
    for (int j = 0; j < 16; ++j)
      a += p[j] * vc[(((size_t)(b * 16 + j) * 16 + h) << 7) + d];
    op[d] = a;
  }
  // importance + stable top-8 (ties -> lower index, like jax.lax.top_k)
  int cur = n >> 6;
  float imp[16];
#pragma unroll
  for (int j = 0; j < 16; ++j) imp[j] = p[j] + (j == cur ? 2.0f : 0.0f);
  unsigned mask = 0;
#pragma unroll
  for (int j = 0; j < 16; ++j) {
    int rank = 0;
#pragma unroll
    for (int i = 0; i < 16; ++i)
      if (imp[i] > imp[j] || (imp[i] == imp[j] && i < j)) rank++;
    if (rank < 8) mask |= 1u << j;
  }
  sel[((size_t)(b * 16 + h) << 10) + n] = mask;
}

// ---------------------------------------------------------------------------
// Fused selected-block + sliding-window attention, one (b,h,16-row q tile)
// per workgroup. 8 waves. S row-block (16x1024 f32) lives in 64KB LDS, then
// is overwritten in place by packed {f16 p_slc, f16 p_swa} probabilities.
// ---------------------------------------------------------------------------
__global__ __launch_bounds__(256) void nsa_attn_kernel(const _Float16* __restrict__ qh,
                                                       const _Float16* __restrict__ kh,
                                                       const _Float16* __restrict__ vt,
                                                       const float* __restrict__ ocmp,
                                                       const float* __restrict__ g,
                                                       const unsigned* __restrict__ sel,
                                                       _Float16* __restrict__ oh) {
  __shared__ unsigned S[16 * 1024];             // 64 KB
  int wg = blockIdx.x;
  int qt = wg & 63, h = (wg >> 6) & 15, b = wg >> 10;
  int qbase = qt << 4;
  int tid = threadIdx.x, wave = tid >> 5, lane = tid & 31;
  int r = lane & 15, hi = lane >> 4;

  // ---- Pass 1: S = scale * Q K^T ; each wave covers 128 keys ----
  const _Float16* qrow = qh + (((size_t)(b * 1024 + qbase + r) * 16 + h) << 7);
  for (int kt = 0; kt < 8; ++kt) {
    int key0 = wave * 128 + kt * 16;
    v8f acc = {0.f,0.f,0.f,0.f,0.f,0.f,0.f,0.f};
    const _Float16* krow = kh + (((size_t)(b * 1024 + key0 + r) * 16 + h) << 7) + hi * 16;
#pragma unroll
    for (int kc = 0; kc < 128; kc += 32) {
      v16h Ao;
      ((v8h*)&Ao)[0] = *(const v8h*)(qrow + kc + hi * 16);
      ((v8h*)&Ao)[1] = *(const v8h*)(qrow + kc + hi * 16 + 8);
      v16h Bo = *(const v16h*)(krow + kc);
      acc = __builtin_amdgcn_wmma_f32_16x16x32_f16(false, Ao, false, Bo, (short)0, acc, false, false);
    }
#pragma unroll
    for (int j = 0; j < 8; ++j)
      S[(j + 8 * hi) * 1024 + key0 + r] = __builtin_bit_cast(unsigned, acc[j] * SCALE);
  }
  __syncthreads();

  // ---- Pass 2: dual masked softmax, pack normalized P pairs in place ----
  {
    int m = tid >> 4, sub = tid & 15;
    int n = qbase + m;
    unsigned bm = sel[((size_t)(b * 16 + h) << 10) + n];
    float mxs = -1e30f, mxw = -1e30f;
    for (int key = sub; key < 1024; key += 16) {
      float s = __builtin_bit_cast(float, S[m * 1024 + key]);
      bool causal = key <= n;
      if (causal && ((bm >> (key >> 6)) & 1)) mxs = fmaxf(mxs, s);
      if (causal && (key >= n - 64))          mxw = fmaxf(mxw, s);
    }
#pragma unroll
    for (int off = 8; off; off >>= 1) {
      mxs = fmaxf(mxs, __shfl_xor(mxs, off, 16));
      mxw = fmaxf(mxw, __shfl_xor(mxw, off, 16));
    }
    float sums = 0.f, sumw = 0.f;
    for (int key = sub; key < 1024; key += 16) {
      float s = __builtin_bit_cast(float, S[m * 1024 + key]);
      bool causal = key <= n;
      float ps = (causal && ((bm >> (key >> 6)) & 1)) ? __expf(s - mxs) : 0.f;
      float pw = (causal && (key >= n - 64))          ? __expf(s - mxw) : 0.f;
      sums += ps; sumw += pw;
      unsigned lo = (unsigned)__builtin_bit_cast(unsigned short, (_Float16)ps);
      unsigned hp = (unsigned)__builtin_bit_cast(unsigned short, (_Float16)pw);
      S[m * 1024 + key] = lo | (hp << 16);
    }
#pragma unroll
    for (int off = 8; off; off >>= 1) {
      sums += __shfl_xor(sums, off, 16);
      sumw += __shfl_xor(sumw, off, 16);
    }
    float invs = sums > 0.f ? 1.0f / sums : 0.f;
    float invw = sumw > 0.f ? 1.0f / sumw : 0.f;
    for (int key = sub; key < 1024; key += 16) {
      unsigned u = S[m * 1024 + key];
      float ps = (float)__builtin_bit_cast(_Float16, (unsigned short)(u & 0xffffu)) * invs;
      float pw = (float)__builtin_bit_cast(_Float16, (unsigned short)(u >> 16)) * invw;
      unsigned lo = (unsigned)__builtin_bit_cast(unsigned short, (_Float16)ps);
      unsigned hp = (unsigned)__builtin_bit_cast(unsigned short, (_Float16)pw);
      S[m * 1024 + key] = lo | (hp << 16);
    }
  }
  __syncthreads();

  // ---- Pass 3: O = P V for both masks; each wave covers 16 channels ----
  int chan0 = wave << 4;
  v8f os = {0.f,0.f,0.f,0.f,0.f,0.f,0.f,0.f};
  v8f ow = os;
  const _Float16* vrow = vt + (((size_t)(b * 16 + h) * 128 + chan0 + r) << 10) + hi * 16;
  const unsigned* Sr = &S[r * 1024];
  for (int kc = 0; kc < 1024; kc += 32) {
    v16h As, Aw;
#pragma unroll
    for (int i = 0; i < 8; ++i) {
      unsigned u = Sr[kc + hi * 16 + i];
      As[i]     = __builtin_bit_cast(_Float16, (unsigned short)(u & 0xffffu));
      Aw[i]     = __builtin_bit_cast(_Float16, (unsigned short)(u >> 16));
      unsigned u2 = Sr[kc + hi * 16 + 8 + i];
      As[8 + i] = __builtin_bit_cast(_Float16, (unsigned short)(u2 & 0xffffu));
      Aw[8 + i] = __builtin_bit_cast(_Float16, (unsigned short)(u2 >> 16));
    }
    v16h Bo = *(const v16h*)(vrow + kc);
    os = __builtin_amdgcn_wmma_f32_16x16x32_f16(false, As, false, Bo, (short)0, os, false, false);
    ow = __builtin_amdgcn_wmma_f32_16x16x32_f16(false, Aw, false, Bo, (short)0, ow, false, false);
  }

  // ---- Pass 4: gate-combine with o_cmp, store f16 o ----
#pragma unroll
  for (int j = 0; j < 8; ++j) {
    int m = j + 8 * hi;
    int n = qbase + m;
    const float* gp = g + (size_t)(b * 1024 + n) * 48;
    float oc = ocmp[(((size_t)(b * 1024 + n) * 16 + h) << 7) + chan0 + r];
    float val = oc * gp[h] + os[j] * gp[16 + h] + ow[j] * gp[32 + h];
    oh[((size_t)(b * 1024 + n) << 11) + h * 128 + chan0 + r] = (_Float16)val;
  }
}

// ---------------------------------------------------------------------------
// Host launcher
// ---------------------------------------------------------------------------
extern "C" void kernel_launch(void* const* d_in, const int* in_sizes, int n_in,
                              void* d_out, int out_size, void* d_ws, size_t ws_size,
                              hipStream_t stream) {
  (void)in_sizes; (void)n_in; (void)out_size; (void)ws_size;
  const float* x  = (const float*)d_in[0];
  const float* Wq = (const float*)d_in[1];
  const float* Wk = (const float*)d_in[2];
  const float* Wv = (const float*)d_in[3];
  const float* Wg = (const float*)d_in[4];
  const float* Wo = (const float*)d_in[5];
  float* out = (float*)d_out;
  char* ws = (char*)d_ws;
  const size_t MB = 1024 * 1024;

  _Float16* xh   = (_Float16*)(ws + 0 * MB);     // 8 MB
  _Float16* wqt  = (_Float16*)(ws + 8 * MB);     // 8 MB
  _Float16* wkt  = (_Float16*)(ws + 16 * MB);    // 8 MB
  _Float16* wvt  = (_Float16*)(ws + 24 * MB);    // 8 MB
  _Float16* wot  = (_Float16*)(ws + 32 * MB);    // 8 MB
  float*    qf   = (float*)   (ws + 40 * MB);    // 16 MB
  float*    kf   = (float*)   (ws + 56 * MB);    // 16 MB
  float*    vf   = (float*)   (ws + 72 * MB);    // 16 MB
  float*    gf   = (float*)   (ws + 88 * MB);    // 384 KB
  _Float16* qhp  = (_Float16*)(ws + 89 * MB);    // 8 MB
  _Float16* khp  = (_Float16*)(ws + 97 * MB);    // 8 MB
  _Float16* vtp  = (_Float16*)(ws + 105 * MB);   // 8 MB
  float*    kcmp = (float*)   (ws + 113 * MB);   // 256 KB
  float*    vcmp = (float*)   (ws + 113 * MB + 256 * 1024); // 256 KB
  float*    ocmp = (float*)   (ws + 114 * MB);   // 16 MB
  unsigned* selp = (unsigned*)(ws + 130 * MB);   // 128 KB
  _Float16* ohp  = (_Float16*)(ws + 131 * MB);   // 8 MB (total 139 MB)

  // 1) precision conversion / weight transposition
  cvt_f32_to_f16<<<16384, 256, 0, stream>>>(x, xh, 2048 * 2048);
  cvt_wt_kernel<<<2048, 256, 0, stream>>>(Wq, wqt, 2048, 2048);
  cvt_wt_kernel<<<2048, 256, 0, stream>>>(Wk, wkt, 2048, 2048);
  cvt_wt_kernel<<<2048, 256, 0, stream>>>(Wv, wvt, 2048, 2048);
  cvt_wt_kernel<<<2048, 256, 0, stream>>>(Wo, wot, 2048, 2048);

  // 2) q/k/v projections (WMMA GEMM) + gates
  gemm_f16_16x64<<<512, 256, 0, stream>>>(xh, wqt, qf, 2048, 2048, 2048);
  gemm_f16_16x64<<<512, 256, 0, stream>>>(xh, wkt, kf, 2048, 2048, 2048);
  gemm_f16_16x64<<<512, 256, 0, stream>>>(xh, wvt, vf, 2048, 2048, 2048);
  gate_kernel<<<384, 256, 0, stream>>>(x, Wg, gf);

  // 3) RoPE + layout repacks + compression
  rope_kernel<<<8192, 256, 0, stream>>>(qf, kf, qhp, khp);
  repack_v_kernel<<<16384, 256, 0, stream>>>(vf, vtp);
  kv_cmp_kernel<<<256, 256, 0, stream>>>(kf, vf, kcmp, vcmp);

  // 4) compressed attention + block selection
  cmp_attn_kernel<<<128, 256, 0, stream>>>(qf, kcmp, vcmp, ocmp, selp);

  // 5) fused slc+swa attention with gated combine -> f16 o
  nsa_attn_kernel<<<2048, 256, 0, stream>>>(qhp, khp, vtp, ocmp, gf, selp, ohp);

  // 6) output projection
  gemm_f16_16x64<<<512, 256, 0, stream>>>(ohp, wot, out, 2048, 2048, 2048);
}
